// GCNGraphClassifier_8332236554725
// MI455X (gfx1250) — compile-verified, matrix operand order checked by reading, not run on hardware
//
#include <hip/hip_runtime.h>

#define NN 50000
#define EE 600000
#define DD 128
#define HH 128
#define CC 16
#define GG 128

typedef __attribute__((ext_vector_type(16))) __bf16 v16bf;
typedef __attribute__((ext_vector_type(8)))  float  v8f;
typedef __attribute__((ext_vector_type(4)))  unsigned int v4u;
typedef __attribute__((ext_vector_type(8)))  int    v8i_t;
typedef __attribute__((ext_vector_type(4)))  int    v4i_t;

__device__ __forceinline__ unsigned short f2bf(float f) {
    unsigned int u = __float_as_uint(f);
    unsigned int r = (u + 0x7FFFu + ((u >> 16) & 1u)) >> 16;  // RNE
    return (unsigned short)r;
}

// ---------------- utility kernels ----------------

__global__ void k_zero(float* p, int n) {
    int i = blockIdx.x * blockDim.x + threadIdx.x;
    if (i < n) p[i] = 0.0f;
}

__global__ void k_deg(const long long* __restrict__ ei, const float* __restrict__ ew,
                      float* __restrict__ deg) {
    int i = blockIdx.x * blockDim.x + threadIdx.x;
    if (i < EE) {
        atomicAdd(&deg[(int)ei[(size_t)EE + i]], ew[i]);      // dst degrees
    } else if (i < EE + NN) {
        atomicAdd(&deg[i - EE], 1.0f);                        // self loop weight 1
    }
}

__global__ void k_dinv(const float* __restrict__ deg, float* __restrict__ dinv) {
    int i = blockIdx.x * blockDim.x + threadIdx.x;
    if (i < NN) { float d = deg[i]; dinv[i] = d > 0.0f ? rsqrtf(d) : 0.0f; }
}

__global__ void k_norm(const long long* __restrict__ ei, const float* __restrict__ ew,
                       const float* __restrict__ dinv, float* __restrict__ norm) {
    int e = blockIdx.x * blockDim.x + threadIdx.x;
    if (e < EE) {
        int s = (int)ei[e], d = (int)ei[(size_t)EE + e];
        norm[e] = dinv[s] * ew[e] * dinv[d];
    } else if (e < EE + NN) {
        int n = e - EE;
        norm[e] = dinv[n] * dinv[n];
    }
}

__global__ void k_cvt_bf16(const float* __restrict__ src, unsigned short* __restrict__ dst, int n) {
    int i = blockIdx.x * blockDim.x + threadIdx.x;
    if (i < n) dst[i] = f2bf(src[i]);
}

// transpose+convert three 128x128 fp32 weights -> bf16 Wt[n][k] = W[k][n]
__global__ void k_prepw(const float* __restrict__ W1, const float* __restrict__ W2,
                        const float* __restrict__ L1, unsigned short* __restrict__ W1t,
                        unsigned short* __restrict__ W2t, unsigned short* __restrict__ L1t) {
    int i = blockIdx.x * blockDim.x + threadIdx.x;
    if (i >= 3 * 128 * 128) return;
    int w = i >> 14, idx = i & 16383;
    int k = idx >> 7, n = idx & 127;
    const float* s = (w == 0) ? W1 : (w == 1) ? W2 : L1;
    unsigned short* d = (w == 0) ? W1t : (w == 1) ? W2t : L1t;
    d[n * 128 + k] = f2bf(s[k * 128 + n]);
}

// ---------------- WMMA GEMM: C[M x 128] = A[M x 128] @ W (Wt is 128x128 transposed bf16)
__global__ __launch_bounds__(256)
void k_gemm128(const unsigned short* __restrict__ A, const unsigned short* __restrict__ Wt,
               float* __restrict__ C, int m_tiles, const float* __restrict__ bias, int do_relu) {
    __shared__ unsigned short ldsW[128 * 128];

    // Stage the whole 32KB weight tile into LDS with one TDM descriptor (wave 0 issues).
    if ((threadIdx.x >> 5) == 0) {
        unsigned lds_addr = (unsigned)(uintptr_t)(&ldsW[0]);       // LDS aperture: offset in low 32b
        unsigned long long ga = (unsigned long long)(uintptr_t)Wt; // 57-bit global byte address
        // D# group0: count=1 | lds_addr | global_addr[56:0] | type=2
        v4u g0;
        g0[0] = 1u;
        g0[1] = lds_addr;
        g0[2] = (unsigned)(ga & 0xFFFFFFFFu);
        g0[3] = (unsigned)((ga >> 32) & 0x01FFFFFFu) | 0x80000000u;
        // D# group1: data_size=1 (2B) | tensor_dim0=128 | tensor_dim1=128 |
        //            tile_dim0=128 | tile_dim1=128 | tensor_dim0_stride=128
        v8i_t g1;
        g1[0] = 0x00010000;   // wg_mask=0, data_size=1 (2 bytes)
        g1[1] = 128 << 16;    // tensor_dim0[15:0] in bits[63:48]
        g1[2] = 128 << 16;    // tensor_dim1[15:0] in bits[95:80]
        g1[3] = 128 << 16;    // tile_dim0 in bits[127:112]
        g1[4] = 128;          // tile_dim1 in bits[143:128]
        g1[5] = 128;          // tensor_dim0_stride[31:0]
        g1[6] = 0;
        g1[7] = 0;
        v4i_t gz4 = {0, 0, 0, 0};          // 2D tensor: groups 2/3 unused
        v8i_t gz8 = {0, 0, 0, 0, 0, 0, 0, 0};
        __builtin_amdgcn_tensor_load_to_lds(g0, g1, gz4, gz4, gz8, 0);
        __builtin_amdgcn_s_wait_tensorcnt(0);
    }
    __syncthreads();

    const int wave = threadIdx.x >> 5;
    const int lane = threadIdx.x & 31;
    const int rtile = blockIdx.x * 8 + wave;
    if (rtile >= m_tiles) return;            // uniform per wave: EXEC stays all-ones inside

    const int r16   = lane & 15;             // row within tile (A), col within tile (B/D)
    const int khalf = lane >> 4;             // K half select: k = 16*khalf + s (contiguous, same for A and B)
    const size_t gr = (size_t)(rtile * 16 + r16);

    v16bf afrag[4];
#pragma unroll
    for (int kt = 0; kt < 4; ++kt)
        afrag[kt] = *(const v16bf*)(A + gr * 128 + kt * 32 + khalf * 16);

#pragma unroll
    for (int ct = 0; ct < 8; ++ct) {
        const int n = ct * 16 + r16;
        v8f acc = {};
#pragma unroll
        for (int kt = 0; kt < 4; ++kt) {
            v16bf bfrag = *(const v16bf*)(ldsW + n * 128 + kt * 32 + khalf * 16);
            acc = __builtin_amdgcn_wmma_f32_16x16x32_bf16(
                false, afrag[kt], false, bfrag, (short)0, acc, false, false);
        }
        const float bv = bias ? bias[n] : 0.0f;
#pragma unroll
        for (int v = 0; v < 8; ++v) {
            float val = acc[v] + bv;
            if (do_relu) val = fmaxf(val, 0.0f);
            C[(size_t)(rtile * 16 + v + 8 * khalf) * 128 + n] = val;  // D layout: M=v+8*khalf
        }
    }
}

// ---------------- edge scatter: Aout[dst] += norm[e] * Hin[src], 4 channels/thread
__global__ void k_aggregate(const float* __restrict__ Hin, const long long* __restrict__ ei,
                            const float* __restrict__ norm, float* __restrict__ Aout) {
    int tid = blockIdx.x * blockDim.x + threadIdx.x;
    int e = tid >> 5;
    if (e >= EE + NN) return;
    int c = (tid & 31) * 4;
    int s, d;
    if (e < EE) { s = (int)ei[e]; d = (int)ei[(size_t)EE + e]; }
    else        { s = d = e - EE; }
    float nm = norm[e];
    float4 h4 = *(const float4*)(Hin + (size_t)s * 128 + c);
    float* dp = Aout + (size_t)d * 128 + c;
    atomicAdd(dp + 0, nm * h4.x);
    atomicAdd(dp + 1, nm * h4.y);
    atomicAdd(dp + 2, nm * h4.z);
    atomicAdd(dp + 3, nm * h4.w);
}

// bias (+ optional relu); writes bf16 copy when bfout != null, else updates in place
__global__ void k_finalize(float* __restrict__ A, const float* __restrict__ bias,
                           int do_relu, unsigned short* __restrict__ bfout, int n) {
    int i = blockIdx.x * blockDim.x + threadIdx.x;
    if (i >= n) return;
    float v = A[i] + bias[i & 127];
    if (do_relu) v = fmaxf(v, 0.0f);
    if (bfout) bfout[i] = f2bf(v);
    else       A[i] = v;
}

__global__ void k_pool(const float* __restrict__ A, const long long* __restrict__ batch,
                       float* __restrict__ gsum) {
    int i = blockIdx.x * blockDim.x + threadIdx.x;
    if (i >= NN * 128) return;
    int node = i >> 7, c = i & 127;
    atomicAdd(&gsum[(int)batch[node] * 128 + c], A[i]);
}

__global__ void k_cnt(const long long* __restrict__ batch, float* __restrict__ gcnt) {
    int i = blockIdx.x * blockDim.x + threadIdx.x;
    if (i < NN) atomicAdd(&gcnt[(int)batch[i]], 1.0f);
}

__global__ void k_gfin(const float* __restrict__ gsum, const float* __restrict__ gcnt,
                       unsigned short* __restrict__ gb) {
    int i = blockIdx.x * blockDim.x + threadIdx.x;
    if (i >= GG * 128) return;
    gb[i] = f2bf(gsum[i] / fmaxf(gcnt[i >> 7], 1.0f));
}

// LayerNorm + final 128->16 linear, one block (128 threads) per graph
__global__ __launch_bounds__(128)
void k_head(const float* __restrict__ z, const float* __restrict__ gamma,
            const float* __restrict__ beta, const float* __restrict__ W2,
            const float* __restrict__ b2, float* __restrict__ out) {
    __shared__ float red[128];
    __shared__ float s[128];
    int g = blockIdx.x, c = threadIdx.x;
    float v = z[g * 128 + c];
    red[c] = v; __syncthreads();
    for (int o = 64; o > 0; o >>= 1) { if (c < o) red[c] += red[c + o]; __syncthreads(); }
    float mu = red[0] / 128.0f; __syncthreads();
    red[c] = v * v; __syncthreads();
    for (int o = 64; o > 0; o >>= 1) { if (c < o) red[c] += red[c + o]; __syncthreads(); }
    float var = red[0] / 128.0f - mu * mu;
    float zn = (v - mu) * rsqrtf(var + 1e-5f) * gamma[c] + beta[c];
    s[c] = zn; __syncthreads();
    if (c < CC) {
        float acc = b2[c];
        for (int j = 0; j < 128; ++j) acc += s[j] * W2[j * CC + c];
        out[g * CC + c] = acc;
    }
}

// ---------------- host side ----------------

static inline size_t alignup(size_t v) { return (v + 255) & ~(size_t)255; }

extern "C" void kernel_launch(void* const* d_in, const int* in_sizes, int n_in,
                              void* d_out, int out_size, void* d_ws, size_t ws_size,
                              hipStream_t stream) {
    const float*     x     = (const float*)d_in[0];
    const long long* ei    = (const long long*)d_in[1];   // int64 [2,E]
    const float*     ew    = (const float*)d_in[2];
    const long long* batch = (const long long*)d_in[3];   // int64 [N]
    const float* W1 = (const float*)d_in[4],  *b1 = (const float*)d_in[5];
    const float* W2 = (const float*)d_in[6],  *b2 = (const float*)d_in[7];
    const float* L1W = (const float*)d_in[8], *L1b = (const float*)d_in[9];
    const float* lng = (const float*)d_in[10], *lnb = (const float*)d_in[11];
    const float* L2W = (const float*)d_in[12], *L2b = (const float*)d_in[13];
    float* out = (float*)d_out;

    char* ws = (char*)d_ws;
    size_t off = 0;
    auto alloc = [&](size_t bytes) { void* p = ws + off; off = alignup(off + bytes); return p; };

    float*          deg  = (float*)alloc(NN * 4);
    float*          dinv = (float*)alloc(NN * 4);
    float*          nrm  = (float*)alloc((size_t)(EE + NN) * 4);
    unsigned short* Xb   = (unsigned short*)alloc((size_t)NN * 128 * 2);
    unsigned short* W1t  = (unsigned short*)alloc(128 * 128 * 2);
    unsigned short* W2t  = (unsigned short*)alloc(128 * 128 * 2);
    unsigned short* L1t  = (unsigned short*)alloc(128 * 128 * 2);
    float*          Hbuf = (float*)alloc((size_t)NN * 128 * 4);
    float*          Abuf = (float*)alloc((size_t)NN * 128 * 4);
    float*          gsum = (float*)alloc(GG * 128 * 4);
    float*          gcnt = (float*)alloc(GG * 4);
    unsigned short* gb   = (unsigned short*)alloc(GG * 128 * 2);
    float*          zbuf = (float*)alloc(GG * 128 * 4);
    (void)ws_size; (void)n_in; (void)in_sizes; (void)out_size;

    const int T = 256;
    const int nTiles = NN / 16;                 // 3125 row tiles
    const int gemmBlocks = (nTiles + 7) / 8;    // 8 waves per block

    // --- gcn_norm ---
    k_zero<<<(NN + T - 1) / T, T, 0, stream>>>(deg, NN);
    k_deg<<<(EE + NN + T - 1) / T, T, 0, stream>>>(ei, ew, deg);
    k_dinv<<<(NN + T - 1) / T, T, 0, stream>>>(deg, dinv);
    k_norm<<<(EE + NN + T - 1) / T, T, 0, stream>>>(ei, ew, dinv, nrm);

    // --- precision prep ---
    k_cvt_bf16<<<((NN * 128) + T - 1) / T, T, 0, stream>>>(x, Xb, NN * 128);
    k_prepw<<<(3 * 128 * 128 + T - 1) / T, T, 0, stream>>>(W1, W2, L1W, W1t, W2t, L1t);

    // --- conv1: transform (WMMA + TDM), scatter, bias+relu ---
    k_gemm128<<<gemmBlocks, T, 0, stream>>>(Xb, W1t, Hbuf, nTiles, nullptr, 0);
    k_zero<<<((NN * 128) + T - 1) / T, T, 0, stream>>>(Abuf, NN * 128);
    k_aggregate<<<(((EE + NN) * 32) + T - 1) / T, T, 0, stream>>>(Hbuf, ei, nrm, Abuf);
    k_finalize<<<((NN * 128) + T - 1) / T, T, 0, stream>>>(Abuf, b1, 1, Xb, NN * 128);

    // --- conv2 ---
    k_gemm128<<<gemmBlocks, T, 0, stream>>>(Xb, W2t, Hbuf, nTiles, nullptr, 0);
    k_zero<<<((NN * 128) + T - 1) / T, T, 0, stream>>>(Abuf, NN * 128);
    k_aggregate<<<(((EE + NN) * 32) + T - 1) / T, T, 0, stream>>>(Hbuf, ei, nrm, Abuf);
    k_finalize<<<((NN * 128) + T - 1) / T, T, 0, stream>>>(Abuf, b2, 0, nullptr, NN * 128);

    // --- global mean pool ---  (gsum and gcnt are contiguous: zero both)
    k_zero<<<((GG * 128 + GG) + T - 1) / T, T, 0, stream>>>(gsum, GG * 128 + GG);
    k_pool<<<((NN * 128) + T - 1) / T, T, 0, stream>>>(Abuf, batch, gsum);
    k_cnt<<<(NN + T - 1) / T, T, 0, stream>>>(batch, gcnt);
    k_gfin<<<((GG * 128) + T - 1) / T, T, 0, stream>>>(gsum, gcnt, gb);

    // --- head: Linear+ReLU (WMMA), LayerNorm + final linear ---
    k_gemm128<<<1, T, 0, stream>>>(gb, L1t, zbuf, GG / 16, L1b, 1);
    k_head<<<GG, 128, 0, stream>>>(zbuf, lng, lnb, L2W, L2b, out);
}
